// NNUE_80092550135974
// MI455X (gfx1250) — matrix operand order repaired and checked by reading, not compiled, not run on hardware
//
#include <hip/hip_runtime.h>

// ---------------------------------------------------------------------------
// NNUE forward for MI455X (gfx1250, wave32, WMMA).
//   w_ = [white,black]@Ww + bw ; b_ = [black,white]@Wb + bb   (the 193-GFLOP part)
//   base = relu(pov ? [w_,b_] : [b_,w_]) ; out = base@Ws + relu-MLP(base)
// Strategy: dual-fused bf16 WMMA GEMM (activations read once), pre-swizzled
// bf16 weight fragments resident in L2, K-split with f32 atomics, VALU tail.
// ---------------------------------------------------------------------------

typedef __attribute__((ext_vector_type(16))) __bf16 v16bf;
typedef __attribute__((ext_vector_type(8)))  float  v8f;

#define HALFK   20480
#define INK     40960
#define BASEN   288
#define NT      18              // 288 / 16 N-tiles
#define KTILE   32              // bf16 WMMA K
#define MT      64              // batch rows per workgroup
#define KSEG    8               // K split for occupancy
#define KT_HALF (HALFK / KTILE) // 640
#define KT_FULL (INK / KTILE)   // 1280
#define BATCH   4096

union Frag { uint4 q[2]; v16bf v; };

static __device__ __forceinline__ unsigned int f2bf(float f) {
  unsigned int u = __float_as_uint(f);
  return (u + 0x7fffu + ((u >> 16) & 1u)) >> 16;   // round-to-nearest-even
}

// ---------------------------------------------------------------------------
// Kernel 1: fp32 weights -> bf16 WMMA B-fragment swizzle.
// Tile (kt, nt) is 1024B: u32 slot (lane, v) at lane*32 + v*4 holds
// W[kt*32 + (lane>=16 ? 16 : 0) + 2v + {0,1}, nt*16 + lane%16]  (K pair packed)
// ---------------------------------------------------------------------------
__global__ __launch_bounds__(256) void nnue_wconv(
    const float* __restrict__ Ww, const float* __restrict__ Wb,
    unsigned int* __restrict__ wwf, unsigned int* __restrict__ wbf)
{
  int t = blockIdx.x * blockDim.x + threadIdx.x;
  const int per = KT_FULL * NT * 256;
  const float* S = Ww;
  unsigned int* D = wwf;
  if (t >= per) { t -= per; S = Wb; D = wbf; }
  const int q    = t & 255;
  const int tile = t >> 8;
  const int nt   = tile % NT;
  const int kt   = tile / NT;
  const int ln   = q >> 3;
  const int v    = q & 7;
  const int k = kt * KTILE + ((ln & 16) ? 16 : 0) + 2 * v;
  const int n = nt * 16 + (ln & 15);
  const float lo = S[(size_t)k * BASEN + n];
  const float hi = S[(size_t)(k + 1) * BASEN + n];
  D[t] = f2bf(lo) | (f2bf(hi) << 16);
}

// ---------------------------------------------------------------------------
// Kernel 2: seed accumulator buffers with the biases (atomics add partials).
// ---------------------------------------------------------------------------
__global__ __launch_bounds__(256) void nnue_binit(
    const float* __restrict__ bw, const float* __restrict__ bb,
    float* __restrict__ wbuf, float* __restrict__ bbuf)
{
  const int t = blockIdx.x * blockDim.x + threadIdx.x;
  const int n = t % BASEN;
  wbuf[t] = bw[n];
  bbuf[t] = bb[n];
}

// ---------------------------------------------------------------------------
// Kernel 3: dual-fused GEMM. 256 threads = 8 waves. Waves 0-3: w_, 4-7: b_.
// Each wave: 16 rows x 288 cols (18 f32 16x16 accumulators = 144 VGPRs).
// Per K-step: white & black tiles staged to LDS as bf16; 36 WMMAs/wave.
// ---------------------------------------------------------------------------
__global__ __launch_bounds__(256, 1) void nnue_gemm(
    const float* __restrict__ white, const float* __restrict__ black,
    const unsigned int* __restrict__ wwf, const unsigned int* __restrict__ wbf,
    float* __restrict__ wbuf, float* __restrict__ bbuf)
{
  __shared__ alignas(16) unsigned short shw[MT * KTILE];   // 4 KB
  __shared__ alignas(16) unsigned short shb[MT * KTILE];   // 4 KB

  const int tid  = threadIdx.x;
  const int lane = tid & 31;
  const int wv   = tid >> 5;
  const int mt   = wv & 3;           // M row-tile within the WG for this wave
  const bool doB = wv >= 4;          // waves 4..7 compute b_
  const int m0   = blockIdx.x * MT;
  const int ksteps = HALFK / KTILE / KSEG;        // 80
  const int ktbase = blockIdx.y * ksteps;

  const v8f vzero = {0.f, 0.f, 0.f, 0.f, 0.f, 0.f, 0.f, 0.f};
  v8f acc[NT];
#pragma unroll
  for (int i = 0; i < NT; ++i) acc[i] = vzero;

  // staging map: thread -> (row 0..63, col-group of 8)
  const int srow = tid >> 2;
  const int scol = (tid & 3) * 8;
  const float* gw = white + (size_t)(m0 + srow) * HALFK + scol;
  const float* gb = black + (size_t)(m0 + srow) * HALFK + scol;

  // A-fragment LDS byte offset: lanes 0-15 read K0-15, 16-31 read K16-31
  const int aoff = (mt * 16 + (lane & 15)) * (KTILE * 2) + ((lane & 16) ? 32 : 0);

  const unsigned int* WF = doB ? wbf : wwf;
  const int ktA0 = doB ? KT_HALF : 0;  // weight half paired with WHITE
  const int ktB0 = doB ? 0 : KT_HALF;  // weight half paired with BLACK

  for (int ks = 0; ks < ksteps; ++ks) {
    const int kt = ktbase + ks;
    __syncthreads();
    {  // stage fp32 -> bf16 tiles into LDS (64x32 each)
      const float4* pw = (const float4*)(gw + (size_t)kt * KTILE);
      const float4* pb = (const float4*)(gb + (size_t)kt * KTILE);
      float4 w0 = pw[0], w1 = pw[1];
      float4 k0 = pb[0], k1 = pb[1];
      uint4 pkw, pkb;
      pkw.x = f2bf(w0.x) | (f2bf(w0.y) << 16);
      pkw.y = f2bf(w0.z) | (f2bf(w0.w) << 16);
      pkw.z = f2bf(w1.x) | (f2bf(w1.y) << 16);
      pkw.w = f2bf(w1.z) | (f2bf(w1.w) << 16);
      pkb.x = f2bf(k0.x) | (f2bf(k0.y) << 16);
      pkb.y = f2bf(k0.z) | (f2bf(k0.w) << 16);
      pkb.z = f2bf(k1.x) | (f2bf(k1.y) << 16);
      pkb.w = f2bf(k1.z) | (f2bf(k1.w) << 16);
      *(uint4*)&shw[srow * KTILE + scol] = pkw;
      *(uint4*)&shb[srow * KTILE + scol] = pkb;
    }
    __syncthreads();

    Frag aw, ab;
    {
      const char* pw = (const char*)shw + aoff;
      const char* pb = (const char*)shb + aoff;
      aw.q[0] = *(const uint4*)(pw);
      aw.q[1] = *(const uint4*)(pw + 16);
      ab.q[0] = *(const uint4*)(pb);
      ab.q[1] = *(const uint4*)(pb + 16);
    }

    const unsigned int* baseA = WF + (size_t)(ktA0 + kt) * NT * 256 + lane * 8;
    const unsigned int* baseB = WF + (size_t)(ktB0 + kt) * NT * 256 + lane * 8;
#pragma unroll
    for (int nt = 0; nt < NT; ++nt) {
      Frag bA, bB;
      bA.q[0] = *(const uint4*)(baseA + nt * 256);
      bA.q[1] = *(const uint4*)(baseA + nt * 256 + 4);
      acc[nt] = __builtin_amdgcn_wmma_f32_16x16x32_bf16(
          false, aw.v, false, bA.v, (short)0, acc[nt], false, false);
      bB.q[0] = *(const uint4*)(baseB + nt * 256);
      bB.q[1] = *(const uint4*)(baseB + nt * 256 + 4);
      acc[nt] = __builtin_amdgcn_wmma_f32_16x16x32_bf16(
          false, ab.v, false, bB.v, (short)0, acc[nt], false, false);
    }
    if (ks + 1 < ksteps) {     // pull next weight K-tiles toward L2/L1
      __builtin_prefetch(baseA + NT * 256, 0, 1);
      __builtin_prefetch(baseB + NT * 256, 0, 1);
    }
  }

  // epilogue: C layout — vgpr r, lanes 0-15: M=r, lanes 16-31: M=8+r, N=lane%16
  float* obuf = doB ? bbuf : wbuf;
  const int row0 = m0 + mt * 16 + ((lane & 16) ? 8 : 0);
  const int col  = lane & 15;
#pragma unroll
  for (int nt = 0; nt < NT; ++nt) {
#pragma unroll
    for (int r = 0; r < 8; ++r) {
      atomicAdd(&obuf[(size_t)(row0 + r) * BASEN + nt * 16 + col], acc[nt][r]);
    }
  }
}

// ---------------------------------------------------------------------------
// Kernel 4: tail. One wave per batch row (4096 waves). pov-select + relu,
// skip dot, 576->32->32->1 MLP via LDS broadcast + shfl reduction.
// ---------------------------------------------------------------------------
__global__ __launch_bounds__(256) void nnue_tail(
    const float* __restrict__ pov,
    const float* __restrict__ wbuf, const float* __restrict__ bbuf,
    const float* __restrict__ Ws, const float* __restrict__ bs,
    const float* __restrict__ W0, const float* __restrict__ b0,
    const float* __restrict__ W1, const float* __restrict__ b1,
    const float* __restrict__ W2, const float* __restrict__ b2,
    float* __restrict__ out)
{
  __shared__ float shbase[8][2 * BASEN];
  __shared__ float shx[8][32];
  const int wv   = threadIdx.x >> 5;
  const int lane = threadIdx.x & 31;
  const int row  = blockIdx.x * 8 + wv;

  const bool p = pov[row] > 0.5f;                      // wave-uniform
  const float* f1 = (p ? wbuf : bbuf) + (size_t)row * BASEN;
  const float* f2 = (p ? bbuf : wbuf) + (size_t)row * BASEN;

  float skip = 0.f;
  for (int i = lane; i < BASEN; i += 32) {
    float v1 = fmaxf(f1[i], 0.f);
    shbase[wv][i] = v1;
    skip = fmaf(v1, Ws[i], skip);
    float v2 = fmaxf(f2[i], 0.f);
    shbase[wv][BASEN + i] = v2;
    skip = fmaf(v2, Ws[BASEN + i], skip);
  }
  __syncthreads();

  float a0 = b0[lane];
  for (int i = 0; i < 2 * BASEN; ++i)
    a0 = fmaf(shbase[wv][i], W0[i * 32 + lane], a0);
  shx[wv][lane] = fmaxf(a0, 0.f);
  __syncthreads();

  float a1 = b1[lane];
#pragma unroll
  for (int i = 0; i < 32; ++i)
    a1 = fmaf(shx[wv][i], W1[i * 32 + lane], a1);

  float part = fmaf(fmaxf(a1, 0.f), W2[lane], skip);
#pragma unroll
  for (int off = 16; off; off >>= 1) part += __shfl_xor(part, off, 32);
  if (lane == 0) out[row] = part + b2[0] + bs[0];
}

// ---------------------------------------------------------------------------
extern "C" void kernel_launch(void* const* d_in, const int* in_sizes, int n_in,
                              void* d_out, int out_size, void* d_ws, size_t ws_size,
                              hipStream_t stream)
{
  (void)in_sizes; (void)n_in; (void)out_size; (void)ws_size;
  const float* pov   = (const float*)d_in[0];
  const float* white = (const float*)d_in[1];
  const float* black = (const float*)d_in[2];
  const float* Ww = (const float*)d_in[3];
  const float* bw = (const float*)d_in[4];
  const float* Wb = (const float*)d_in[5];
  const float* bb = (const float*)d_in[6];
  const float* Ws = (const float*)d_in[7];
  const float* bs = (const float*)d_in[8];
  const float* W0 = (const float*)d_in[9];
  const float* b0 = (const float*)d_in[10];
  const float* W1 = (const float*)d_in[11];
  const float* b1 = (const float*)d_in[12];
  const float* W2 = (const float*)d_in[13];
  const float* b2 = (const float*)d_in[14];
  float* out = (float*)d_out;

  char* ws = (char*)d_ws;
  const size_t FRAG_BYTES = (size_t)KT_FULL * NT * 1024;         // 23.6 MB each
  unsigned int* wwf = (unsigned int*)(ws);
  unsigned int* wbf = (unsigned int*)(ws + FRAG_BYTES);
  float* wbuf = (float*)(ws + 2 * FRAG_BYTES);                   // [4096][288]
  float* bbuf = (float*)(ws + 2 * FRAG_BYTES + (size_t)BATCH * BASEN * 4);

  {
    const int total = 2 * KT_FULL * NT * 256;                    // 11,796,480
    nnue_wconv<<<total / 256, 256, 0, stream>>>(Ww, Wb, wwf, wbf);
  }
  {
    const int total = BATCH * BASEN;                             // 1,179,648
    nnue_binit<<<total / 256, 256, 0, stream>>>(bw, bb, wbuf, bbuf);
  }
  nnue_gemm<<<dim3(BATCH / MT, KSEG), 256, 0, stream>>>(white, black, wwf, wbf, wbuf, bbuf);
  nnue_tail<<<BATCH / 8, 256, 0, stream>>>(pov, wbuf, bbuf, Ws, bs, W0, b0, W1, b1, W2, b2, out);
}
